// AttentionLayer_46136538694082
// MI455X (gfx1250) — matrix operand order repaired and checked by reading, not compiled, not run on hardware
//
#include <hip/hip_runtime.h>
#include <hip/hip_bf16.h>

typedef __attribute__((ext_vector_type(16))) __bf16        v16bf;
typedef __attribute__((ext_vector_type(8)))  float         v8f;
typedef __attribute__((ext_vector_type(8)))  unsigned int  v8u;

#define B_   4096
#define T_   200
#define D_   64
#define H1_  80
#define H2_  40
#define TP   208     // T padded to 13*16
#define K1   128     // GEMM1 K (fact | fact*q)
#define K1U  64      // u32 (bf16-pair) row length of A / W1T
#define H1P  96      // h1 row padded (bf16 units), K for GEMM2
#define H1PU 48      // u32 row length of h1 / W2T
#define H2P  48      // h2 row padded (bf16 units)

// workspace layout (bytes)
#define WS_W1T   0        // bf16 [80][128]  = 20480 B
#define WS_W2T   20480    // bf16 [48][96]   =  9216 B
#define WS_W1QS  29696    // f32  [80][64]   = 20480 B

// LDS layout (bytes)
#define L_A     0         // u32 [208][64]        53248
#define L_H1    53248     // u32 [208][48]        39936
#define L_H2    93184     // u16 [208][48]        19968
#define L_W1T   113152    // u32 [80][64]         20480
#define L_W2T   133632    // u32 [48][48]          9216
#define L_QC    142848    // f32 [80]               320
#define L_B2    143168    // f32 [48]               192
#define L_W3    143360    // f32 [40]               160
#define L_QV    143520    // f32 [64]               256
#define L_SC    143776    // f32 [208]              832
#define L_PR    144608    // f32 [208]              832
#define L_RED   145440    // f32 [256]             1024
#define SMEM_BYTES 146464

__device__ __forceinline__ unsigned short f2bf(float f) {
  unsigned u = __builtin_bit_cast(unsigned, f);
  u += 0x7FFFu + ((u >> 16) & 1u);          // round-to-nearest-even
  return (unsigned short)(u >> 16);
}
__device__ __forceinline__ unsigned pack2bf(float a, float b) {
  return (unsigned)f2bf(a) | ((unsigned)f2bf(b) << 16);
}
__device__ __forceinline__ float bf2f(unsigned short h) {
  unsigned u = ((unsigned)h) << 16;
  return __builtin_bit_cast(float, u);
}
// fast sigmoid: v_exp_f32 + v_rcp_f32 (avoid IEEE div_scale sequence)
__device__ __forceinline__ float sigmoidf(float x) {
  return __builtin_amdgcn_rcpf(1.0f + __expf(-x));
}

// --- prep: fold W1 blocks, convert weights to bf16 (runs once, tiny) ---
__global__ void din_prep_kernel(const float* __restrict__ W1,
                                const float* __restrict__ W2,
                                char* __restrict__ ws) {
  unsigned short* w1t  = (unsigned short*)(ws + WS_W1T);   // [n=80][k=128]
  unsigned short* w2t  = (unsigned short*)(ws + WS_W2T);   // [n=48][k=96]
  float*          w1qs = (float*)(ws + WS_W1QS);           // [n=80][k=64]
  int i = blockIdx.x * blockDim.x + threadIdx.x;
  if (i < H1_ * K1) {                       // W1eff^T: fact coef = W1[0:64]-W1[192:256]; factq coef = W1[128:192]
    int n = i / K1, k = i % K1;
    float v = (k < 64) ? (W1[k * H1_ + n] - W1[(192 + k) * H1_ + n])
                       : W1[(64 + k) * H1_ + n];           // 128 + (k-64)
    w1t[i] = f2bf(v);
  }
  if (i < H2P * H1P) {                      // W2^T zero-padded to [48][96]
    int n = i / H1P, k = i % H1P;
    float v = (n < H2_ && k < H1_) ? W2[k * H2_ + n] : 0.0f;
    w2t[i] = f2bf(v);
  }
  if (i < H1_ * D_) {                       // q-coef: W1[64:128]+W1[192:256], transposed
    int n = i / D_, k = i % D_;
    w1qs[i] = W1[(64 + k) * H1_ + n] + W1[(192 + k) * H1_ + n];
  }
}

// --- main: one workgroup (8 waves) per batch row ---
__global__ __launch_bounds__(256) void din_attn_kernel(
    const float* __restrict__ query, const float* __restrict__ fact,
    const int* __restrict__ mask,    const float* __restrict__ b1g,
    const float* __restrict__ b2g,   const float* __restrict__ w3g,
    const float* __restrict__ b3g,
    const unsigned short* __restrict__ w1t_g,
    const unsigned short* __restrict__ w2t_g,
    const float* __restrict__ w1qs_g, float* __restrict__ out) {
  extern __shared__ char smem[];
  unsigned*       Au   = (unsigned*)(smem + L_A);
  unsigned*       H1u  = (unsigned*)(smem + L_H1);
  unsigned short* H2s  = (unsigned short*)(smem + L_H2);
  unsigned*       W1Tu = (unsigned*)(smem + L_W1T);
  unsigned*       W2Tu = (unsigned*)(smem + L_W2T);
  float* qc   = (float*)(smem + L_QC);
  float* b2p  = (float*)(smem + L_B2);
  float* w3s  = (float*)(smem + L_W3);
  float* qv   = (float*)(smem + L_QV);
  float* scS  = (float*)(smem + L_SC);
  float* prS  = (float*)(smem + L_PR);
  float* red  = (float*)(smem + L_RED);

  const int tid = threadIdx.x;
  const int b   = blockIdx.x;
  const float* factb = fact + (size_t)b * T_ * D_;

  // P0: query vector to LDS
  if (tid < D_) qv[tid] = query[b * D_ + tid];
  __syncthreads();

  // P1: build A = [fact | fact*q] in bf16; stage weights; per-batch bias qc
  for (int i = tid; i < TP * K1U; i += 256) {
    int row = i >> 6, p = i & 63;
    unsigned v = 0u;
    if (row < T_) {
      if (p < 32) {
        v = pack2bf(factb[row * D_ + 2 * p], factb[row * D_ + 2 * p + 1]);
      } else {
        int j = p - 32;
        float f0 = factb[row * D_ + 2 * j], f1 = factb[row * D_ + 2 * j + 1];
        v = pack2bf(f0 * qv[2 * j], f1 * qv[2 * j + 1]);
      }
    }
    Au[i] = v;
  }
  const unsigned* w1u = (const unsigned*)w1t_g;
  for (int i = tid; i < H1_ * K1U; i += 256) W1Tu[i] = w1u[i];
  const unsigned* w2u = (const unsigned*)w2t_g;
  for (int i = tid; i < H2P * H1PU; i += 256) W2Tu[i] = w2u[i];
  for (int i = tid; i < TP * 8; i += 256)                 // zero h1 K-pad cols 80..95
    H1u[(i >> 3) * H1PU + 40 + (i & 7)] = 0u;
  if (tid < H1_) {                                        // qc = q @ (W1b+W1d) + b1
    float s = b1g[tid];
    const float* wr = w1qs_g + tid * D_;
    for (int k = 0; k < D_; ++k) s += qv[k] * wr[k];
    qc[tid] = s;
  }
  if (tid < H2P) b2p[tid] = (tid < H2_) ? b2g[tid] : 0.0f;
  if (tid < H2_) w3s[tid] = w3g[tid];
  __syncthreads();

  const int wave = tid >> 5;
  const int lm   = tid & 15;        // lane % 16
  const int lh   = (tid >> 4) & 1;  // lane half

  // P2: GEMM1  h1 = sigmoid(A[208x128] @ W1eff[128x80] + qc), bf16 WMMA
  for (int tile = wave; tile < 13 * 5; tile += 8) {
    const int mt = tile / 5, nt = tile % 5;
    v8f acc = {};
    const int m = mt * 16 + lm;
    const int n = nt * 16 + lm;
    #pragma unroll
    for (int ks = 0; ks < 4; ++ks) {
      const int kb = ks * 16;                       // u32 units (32 bf16)
      const unsigned* ap = Au + m * K1U + kb + 4 * lh;
      uint4 a0 = *(const uint4*)ap;
      uint4 a1 = *(const uint4*)(ap + 8);
      v8u aw = {a0.x, a0.y, a0.z, a0.w, a1.x, a1.y, a1.z, a1.w};
      const unsigned* bp = W1Tu + n * K1U + kb + 4 * lh;
      uint4 bl0 = *(const uint4*)bp;
      uint4 bl1 = *(const uint4*)(bp + 8);
      v8u bw = {bl0.x, bl0.y, bl0.z, bl0.w, bl1.x, bl1.y, bl1.z, bl1.w};
      acc = __builtin_amdgcn_wmma_f32_16x16x32_bf16(
          false, __builtin_bit_cast(v16bf, aw),
          false, __builtin_bit_cast(v16bf, bw),
          (short)0, acc, false, false);
    }
    const int col = nt * 16 + lm;
    const float qcv = qc[col];
    unsigned short* h1s = (unsigned short*)H1u;
    #pragma unroll
    for (int r = 0; r < 8; ++r) {
      int row = mt * 16 + 8 * lh + r;
      h1s[row * H1P + col] = f2bf(sigmoidf(acc[r] + qcv));
    }
  }
  __syncthreads();

  // P3: GEMM2  h2 = sigmoid(h1[208x96] @ W2[96x48] + b2), bf16 WMMA
  for (int tile = wave; tile < 13 * 3; tile += 8) {
    const int mt = tile / 3, nt = tile % 3;
    v8f acc = {};
    const int m = mt * 16 + lm;
    const int n = nt * 16 + lm;
    #pragma unroll
    for (int ks = 0; ks < 3; ++ks) {
      const int kb = ks * 16;
      const unsigned* ap = H1u + m * H1PU + kb + 4 * lh;
      uint4 a0 = *(const uint4*)ap;
      uint4 a1 = *(const uint4*)(ap + 8);
      v8u aw = {a0.x, a0.y, a0.z, a0.w, a1.x, a1.y, a1.z, a1.w};
      const unsigned* bp = W2Tu + n * H1PU + kb + 4 * lh;
      uint4 bl0 = *(const uint4*)bp;
      uint4 bl1 = *(const uint4*)(bp + 8);
      v8u bw = {bl0.x, bl0.y, bl0.z, bl0.w, bl1.x, bl1.y, bl1.z, bl1.w};
      acc = __builtin_amdgcn_wmma_f32_16x16x32_bf16(
          false, __builtin_bit_cast(v16bf, aw),
          false, __builtin_bit_cast(v16bf, bw),
          (short)0, acc, false, false);
    }
    const int col = nt * 16 + lm;
    const float bb = b2p[col];
    #pragma unroll
    for (int r = 0; r < 8; ++r) {
      int row = mt * 16 + 8 * lh + r;
      H2s[row * H2P + col] = f2bf(sigmoidf(acc[r] + bb));
    }
  }
  __syncthreads();

  // P4: scores (N=1 GEMV) + mask
  if (tid < T_) {
    float s = b3g[0];
    const unsigned short* hr = H2s + tid * H2P;
    for (int k = 0; k < H2_; ++k) s += bf2f(hr[k]) * w3s[k];
    scS[tid] = (mask[b * T_ + tid] == 1) ? s : -2147483648.0f;
  }
  __syncthreads();

  // P5: softmax (max, exp, sum) over T
  red[tid] = (tid < T_) ? scS[tid] : -3.402823466e38f;
  __syncthreads();
  for (int s = 128; s > 0; s >>= 1) {
    if (tid < s) red[tid] = fmaxf(red[tid], red[tid + s]);
    __syncthreads();
  }
  const float mx = red[0];
  __syncthreads();
  float e = (tid < T_) ? __expf(scS[tid] - mx) : 0.0f;
  red[tid] = e;
  __syncthreads();
  for (int s = 128; s > 0; s >>= 1) {
    if (tid < s) red[tid] += red[tid + s];
    __syncthreads();
  }
  const float zr = __builtin_amdgcn_rcpf(red[0]);
  __syncthreads();
  if (tid < T_) {
    float mbit = (mask[b * T_ + tid] == 1) ? 1.0f : 0.0f;
    prS[tid] = e * zr * mbit;
  }
  __syncthreads();

  // P6: out[b,d] = sum_t p[t] * fact[b,t,d]  (f32 from global)
  const int d = tid & 63;
  const int g = tid >> 6;
  float acc = 0.0f;
  for (int t = g; t < T_; t += 4) acc += prS[t] * factb[t * D_ + d];
  red[tid] = acc;
  __syncthreads();
  if (tid < 64)
    out[b * D_ + tid] = red[tid] + red[64 + tid] + red[128 + tid] + red[192 + tid];
}

extern "C" void kernel_launch(void* const* d_in, const int* in_sizes, int n_in,
                              void* d_out, int out_size, void* d_ws, size_t ws_size,
                              hipStream_t stream) {
  const float* query = (const float*)d_in[0];
  const float* fact  = (const float*)d_in[1];
  const int*   mask  = (const int*)d_in[2];
  const float* W1    = (const float*)d_in[3];
  const float* b1    = (const float*)d_in[4];
  const float* W2    = (const float*)d_in[5];
  const float* b2    = (const float*)d_in[6];
  const float* W3    = (const float*)d_in[7];
  const float* b3    = (const float*)d_in[8];
  float* out = (float*)d_out;
  char*  ws  = (char*)d_ws;

  din_prep_kernel<<<40, 256, 0, stream>>>(W1, W2, ws);

  const unsigned short* w1t  = (const unsigned short*)(ws + WS_W1T);
  const unsigned short* w2t  = (const unsigned short*)(ws + WS_W2T);
  const float*          w1qs = (const float*)(ws + WS_W1QS);

  din_attn_kernel<<<B_, 256, SMEM_BYTES, stream>>>(
      query, fact, mask, b1, b2, W3, b3, w1t, w2t, w1qs, out);
}
